// LSTMAttentionForecaster_62955630625435
// MI455X (gfx1250) — matrix operand order repaired and checked by reading, not compiled, not run on hardware
//
#include <hip/hip_runtime.h>
#include <hip/hip_bf16.h>

// ---- types ---------------------------------------------------------------
typedef _Float16 v16h __attribute__((ext_vector_type(16)));
typedef float    v8f  __attribute__((ext_vector_type(8)));
typedef unsigned int u32x4 __attribute__((ext_vector_type(4)));
typedef float    f32x4 __attribute__((ext_vector_type(4)));

#define HH  256
#define II  64
#define OO  24
#define SS  1024
#define LDH 264   // f16 row stride for h_lds (pad -> conflict-free ds_load_b128)
#define LDX 72    // f16 row stride for x_lds
#define LDE 264   // f32 row stride for E/acc

union FragH { u32x4 u[2]; v16h v; };

// Opaque 0 in an SGPR: the compiler cannot prove it constant, so pointers
// biased by it are not loop-invariant -> weight loads stay inside the t-loop
// (L2-resident streaming) instead of being hoisted + spilled to scratch.
__device__ __forceinline__ int opaque_zero() {
  int z;
  asm volatile("s_mov_b32 %0, 0" : "=s"(z));
  return z;
}

// A fragment, 16x32 f16, per ISA layout:
// lane l<16  (m=l)   : k = k0+0..7   and k0+16..23
// lane l>=16 (m=l-16): k = k0+8..15  and k0+24..31
__device__ __forceinline__ v16h load_afrag(const _Float16* base, int m, int k0,
                                           int half, int ldm) {
  FragH f;
  f.u[0] = *reinterpret_cast<const u32x4*>(base + m * ldm + k0 + half * 8);
  f.u[1] = *reinterpret_cast<const u32x4*>(base + m * ldm + k0 + 16 + half * 8);
  return f.v;
}

// B fragment: pre-swizzled in workspace, 512 f16 per fragment, lane-contiguous
// (two global_load_b128 per lane).
__device__ __forceinline__ v16h load_bfrag(const _Float16* frags, int fragidx,
                                           int lane) {
  return *reinterpret_cast<const v16h*>(frags + (size_t)fragidx * 512 + lane * 16);
}

__device__ __forceinline__ float sigmoidf_(float x) {
  return 1.f / (1.f + __expf(-x));
}
__device__ __forceinline__ float tanhf_(float x) {
  x = fminf(15.f, fmaxf(-15.f, x));
  float e = __expf(2.f * x);
  return (e - 1.f) / (e + 1.f);
}

// ---- prep: fp32 weights -> f16 WMMA B-fragment layout in workspace -------
// frag f < 640 : Wcat = [Wi;Wh] (320x1024), tile=f/10, kc=f%10
// frag f >= 640: attn_W (256x256),          tile=(f-640)/8, kc=(f-640)%8
// element e in [0,512): lane=e/16, j=e%16 ; n = tile*16 + (lane&15),
//                       k = kc*32 + (j<8 ? (lane>>4)*8 + j : 16 + (lane>>4)*8 + (j-8))
__global__ __launch_bounds__(256) void pack_weights_kernel(
    const float* __restrict__ Wi, const float* __restrict__ Wh,
    const float* __restrict__ attnW, _Float16* __restrict__ frags) {
  const int f = blockIdx.x;
  const int tid = threadIdx.x;
#pragma unroll
  for (int e2 = 0; e2 < 2; ++e2) {
    const int e = tid * 2 + e2;
    const int lane = e >> 4, j = e & 15;
    const int k_in = (j < 8) ? (lane >> 4) * 8 + j : 16 + (lane >> 4) * 8 + (j - 8);
    float src;
    if (f < 640) {
      const int tile = f / 10, kc = f % 10;
      const int n = tile * 16 + (lane & 15);
      const int k = kc * 32 + k_in;
      src = (k < II) ? Wi[k * (4 * HH) + n] : Wh[(k - II) * (4 * HH) + n];
    } else {
      const int g2 = f - 640;
      const int tile = g2 / 8, kc = g2 % 8;
      const int n = tile * 16 + (lane & 15);
      const int k = kc * 32 + k_in;
      src = attnW[k * HH + n];
    }
    frags[(size_t)f * 512 + e] = (_Float16)src;
  }
}

// ---- fused LSTM + attention (online softmax) + FC ------------------------
// 16 workgroups x 256 threads (8 waves). WG handles batch rows [wg*16, wg*16+16).
// Wave w owns hidden-column tile pair {2w, 2w+1} across all four gates, so the
// gate fusion and cell state c live entirely in registers for 1024 steps.
__global__ __launch_bounds__(256) void lstm_attn_fused_kernel(
    const float* __restrict__ x, const float* __restrict__ b,
    const float* __restrict__ attn_b, const float* __restrict__ v,
    const float* __restrict__ fcW, const float* __restrict__ fcb,
    const _Float16* __restrict__ wcat_frag,   // 640 fragments
    const _Float16* __restrict__ attn_frag,   // 128 fragments
    float* __restrict__ out) {
  __shared__ _Float16 h_lds[16 * LDH];
  __shared__ _Float16 x_lds[16 * LDX];
  __shared__ float E_lds[16 * LDE];
  __shared__ float acc_lds[16 * LDE];
  __shared__ float v_lds[HH];
  __shared__ float score_lds[16], p_lds[16], corr_lds[16], m_lds[16], l_lds[16];

  const int tid = threadIdx.x;
  const int w = tid >> 5, lane = tid & 31;
  const int n = lane & 15, half = lane >> 4;
  const int m0 = blockIdx.x * 16;

  // ---- init ----
  for (int i = tid; i < HH; i += 256) v_lds[i] = v[i];
  for (int i = tid; i < 16 * LDE; i += 256) acc_lds[i] = 0.f;
  for (int i = tid; i < 16 * LDH; i += 256) h_lds[i] = (_Float16)0.f;
  if (tid < 16) { m_lds[tid] = -3.0e38f; l_lds[tid] = 0.f; score_lds[tid] = 0.f; }

  // stage x_0 (16 rows x 64 cols fp32 -> f16 LDS)
  {
    const int m = tid >> 4, i4 = (tid & 15) * 4;
    f32x4 xv = *reinterpret_cast<const f32x4*>(
        x + (size_t)(m0 + m) * SS * II + i4);
    _Float16* d = &x_lds[m * LDX + i4];
    d[0] = (_Float16)xv.x; d[1] = (_Float16)xv.y;
    d[2] = (_Float16)xv.z; d[3] = (_Float16)xv.w;
  }

  // per-lane constants (biases for this lane's output columns)
  float bi[2], bff[2], bg[2], bo[2], ab[2];
#pragma unroll
  for (int p = 0; p < 2; ++p) {
    const int hc = (2 * w + p) * 16 + n;
    bi[p] = b[0 * HH + hc]; bff[p] = b[1 * HH + hc];
    bg[p] = b[2 * HH + hc]; bo[p] = b[3 * HH + hc];
    ab[p] = attn_b[hc];
  }

  float c_reg[2][8];
#pragma unroll
  for (int p = 0; p < 2; ++p)
#pragma unroll
    for (int r = 0; r < 8; ++r) c_reg[p][r] = 0.f;

  for (int t = 0; t < SS; ++t) {
    // Opaque pointer bias: keeps weight loads inside the loop (no LICM/spill).
    const int zz = opaque_zero();
    const _Float16* wf = wcat_frag + zz;
    const _Float16* af = attn_frag + zz;

    __syncthreads();  // x_t staged, h_{t-1} ready, prev acc update done

    // ---- z = [x_t | h] @ [Wi;Wh]  :  [16,320] x [320,1024] ----
    v8f zacc[4][2];
#pragma unroll
    for (int g = 0; g < 4; ++g)
#pragma unroll
      for (int p = 0; p < 2; ++p) zacc[g][p] = (v8f){0, 0, 0, 0, 0, 0, 0, 0};

#pragma unroll
    for (int kc = 0; kc < 10; ++kc) {
      const v16h a = (kc < 2) ? load_afrag(x_lds, n, kc * 32, half, LDX)
                              : load_afrag(h_lds, n, kc * 32 - II, half, LDH);
      // Issue all 8 B-fragment loads for this K-chunk, then the 8 WMMAs.
      v16h bv[4][2];
#pragma unroll
      for (int g = 0; g < 4; ++g)
#pragma unroll
        for (int p = 0; p < 2; ++p) {
          const int tile = g * 16 + 2 * w + p;
          bv[g][p] = load_bfrag(wf, tile * 10 + kc, lane);
        }
#pragma unroll
      for (int g = 0; g < 4; ++g)
#pragma unroll
        for (int p = 0; p < 2; ++p)
          zacc[g][p] = __builtin_amdgcn_wmma_f32_16x16x32_f16(
              false, a, false, bv[g][p], (short)0, zacc[g][p], false, false);
    }

    __syncthreads();  // all waves done reading h_lds / x_lds

    // ---- gates (registers) -> c, h ; write h_t f16 to LDS ----
#pragma unroll
    for (int p = 0; p < 2; ++p) {
      const int hc = (2 * w + p) * 16 + n;
#pragma unroll
      for (int r = 0; r < 8; ++r) {
        const float iv = sigmoidf_(zacc[0][p][r] + bi[p]);
        const float fv = sigmoidf_(zacc[1][p][r] + bff[p]);
        const float gv = tanhf_(zacc[2][p][r] + bg[p]);
        const float ov = sigmoidf_(zacc[3][p][r] + bo[p]);
        const float c = fv * c_reg[p][r] + iv * gv;
        c_reg[p][r] = c;
        h_lds[(r + 8 * half) * LDH + hc] = (_Float16)(ov * tanhf_(c));
      }
    }
    // stage x_{t+1}
    if (t + 1 < SS) {
      const int m = tid >> 4, i4 = (tid & 15) * 4;
      f32x4 xv = *reinterpret_cast<const f32x4*>(
          x + (size_t)(m0 + m) * SS * II + (size_t)(t + 1) * II + i4);
      _Float16* d = &x_lds[m * LDX + i4];
      d[0] = (_Float16)xv.x; d[1] = (_Float16)xv.y;
      d[2] = (_Float16)xv.z; d[3] = (_Float16)xv.w;
    }
    if (tid < 16) score_lds[tid] = 0.f;

    __syncthreads();  // h_t visible

    // ---- energy = tanh(h_t @ attn_W + attn_b) : [16,256] x [256,256] ----
    v8f eacc[2];
#pragma unroll
    for (int p = 0; p < 2; ++p) eacc[p] = (v8f){0, 0, 0, 0, 0, 0, 0, 0};
#pragma unroll
    for (int kc = 0; kc < 8; ++kc) {
      const v16h a = load_afrag(h_lds, n, kc * 32, half, LDH);
      v16h bv[2];
#pragma unroll
      for (int p = 0; p < 2; ++p)
        bv[p] = load_bfrag(af, (2 * w + p) * 8 + kc, lane);
#pragma unroll
      for (int p = 0; p < 2; ++p)
        eacc[p] = __builtin_amdgcn_wmma_f32_16x16x32_f16(
            false, a, false, bv[p], (short)0, eacc[p], false, false);
    }
#pragma unroll
    for (int p = 0; p < 2; ++p) {
      const int hc = (2 * w + p) * 16 + n;
#pragma unroll
      for (int r = 0; r < 8; ++r)
        E_lds[(r + 8 * half) * LDE + hc] = tanhf_(eacc[p][r] + ab[p]);
    }
    __syncthreads();  // E ready, scores zeroed

    // ---- score[m] = energy[m,:] . v ----
    {
      const int row = tid >> 4, seg = tid & 15;
      float s = 0.f;
#pragma unroll
      for (int j = 0; j < 16; ++j)
        s += E_lds[row * LDE + seg * 16 + j] * v_lds[seg * 16 + j];
      atomicAdd(&score_lds[row], s);
    }
    __syncthreads();  // scores ready

    // ---- online softmax scalars ----
    if (tid < 16) {
      const float s = score_lds[tid];
      const float mo = m_lds[tid];
      const float mn = fmaxf(mo, s);
      const float corr = __expf(mo - mn);
      const float pp = __expf(s - mn);
      m_lds[tid] = mn;
      l_lds[tid] = l_lds[tid] * corr + pp;
      corr_lds[tid] = corr;
      p_lds[tid] = pp;
    }
    __syncthreads();  // p, corr ready

    // ---- acc = acc*corr + p * h_t  (streamed context) ----
    {
      const int row = tid >> 4, seg = tid & 15;
      const float corr = corr_lds[row], pp = p_lds[row];
#pragma unroll
      for (int j = 0; j < 16; ++j) {
        const int hc = seg * 16 + j;
        acc_lds[row * LDE + hc] =
            acc_lds[row * LDE + hc] * corr + pp * (float)h_lds[row * LDH + hc];
      }
    }
  }

  __syncthreads();

  // ---- out = (acc/l) @ fc_W + fc_b ----
  for (int idx = tid; idx < 16 * OO; idx += 256) {
    const int m = idx / OO, o = idx % OO;
    const float inv_l = 1.f / l_lds[m];
    float s = fcb[o];
    for (int hc = 0; hc < HH; ++hc)
      s += (acc_lds[m * LDE + hc] * inv_l) * fcW[hc * OO + o];
    out[(size_t)(m0 + m) * OO + o] = s;
  }
}

// ---- launch --------------------------------------------------------------
extern "C" void kernel_launch(void* const* d_in, const int* in_sizes, int n_in,
                              void* d_out, int out_size, void* d_ws,
                              size_t ws_size, hipStream_t stream) {
  const float* x      = (const float*)d_in[0];
  const float* Wi     = (const float*)d_in[1];
  const float* Wh     = (const float*)d_in[2];
  const float* b      = (const float*)d_in[3];
  const float* attn_W = (const float*)d_in[4];
  const float* attn_b = (const float*)d_in[5];
  const float* v      = (const float*)d_in[6];
  const float* fc_W   = (const float*)d_in[7];
  const float* fc_b   = (const float*)d_in[8];
  float* out = (float*)d_out;

  _Float16* frags = (_Float16*)d_ws;       // 768 fragments * 1 KB = 768 KB
  const _Float16* wcat_frag = frags;
  const _Float16* attn_frag = frags + (size_t)640 * 512;

  pack_weights_kernel<<<768, 256, 0, stream>>>(Wi, Wh, attn_W, frags);
  lstm_attn_fused_kernel<<<16, 256, 0, stream>>>(
      x, b, attn_b, v, fc_W, fc_b, wcat_frag, attn_frag, out);
}
